// AttentionDecoder_17763984736505
// MI455X (gfx1250) — compile-verified, matrix operand order checked by reading, not run on hardware
//
#include <hip/hip_runtime.h>
#include <math.h>

// ---------------------------------------------------------------------------
// AttentionDecoder for MI455X (gfx1250): bf16 WMMA GEMMs + L2-resident bf16
// K/V/logitK caches + fused per-step attention/softmax/argmax kernels.
// ---------------------------------------------------------------------------

typedef __attribute__((ext_vector_type(16))) __bf16 v16bf;
typedef __attribute__((ext_vector_type(8)))  float  v8f;
typedef unsigned short ushort_t;

#define BB 256
#define NN 512
#define DD 128
#define HH 8
#define DKV 16
#define NEGV (-1e9f)
#define CLIPV 10.0f

// ---- bf16 helpers ---------------------------------------------------------
__device__ __forceinline__ float bf2f(unsigned u16) {
    unsigned u = (u16 & 0xffffu) << 16;
    return __builtin_bit_cast(float, u);
}
__device__ __forceinline__ ushort_t f2bf(float f) {
    unsigned u = __builtin_bit_cast(unsigned, f);
    unsigned r = u + 0x7fffu + ((u >> 16) & 1u);   // round-to-nearest-even
    return (ushort_t)(r >> 16);
}
__device__ __forceinline__ void unpack8(uint4 p, float* f) {
    f[0] = bf2f(p.x); f[1] = bf2f(p.x >> 16);
    f[2] = bf2f(p.y); f[3] = bf2f(p.y >> 16);
    f[4] = bf2f(p.z); f[5] = bf2f(p.z >> 16);
    f[6] = bf2f(p.w); f[7] = bf2f(p.w >> 16);
}

// WMMA 16x32 bf16 A/B lane->K mapping (ISA 7.12.2):
// lanes 0-15 hold K {0..7,16..23}; lanes 16-31 hold K {8..15,24..31}.
__device__ __forceinline__ int krel_map(int lane, int i) {
    int base = (lane & 16) ? 8 : 0;
    return base + ((i < 8) ? i : i + 8);
}

union Frag32B { uint4 q[2]; v16bf v; };

// ---- init: visited / current (must reset every launch) --------------------
__global__ void k_init(unsigned char* visited, int* current) {
    int i = blockIdx.x * blockDim.x + threadIdx.x;
    if (i < BB * NN) visited[i] = 0;
    if (i < BB) current[i] = 0;
}

// ---- repack W_node and W_out into WMMA B-fragment layout (bf16) -----------
// frag[((kt*NT + tn)*32 + lane)*16 + i] = W[(kt*32 + krel(lane,i)) * width + tn*16 + (lane&15)]
__global__ void k_weight_frags(const float* __restrict__ Wnode,
                               const float* __restrict__ Wout,
                               ushort_t* __restrict__ wnode_frag,
                               ushort_t* __restrict__ wout_frag) {
    int idx = blockIdx.x * blockDim.x + threadIdx.x;
    const int NODE_TOT = 4 * 24 * 32 * 16;   // ktiles * coltiles * lanes * elems
    const int OUT_TOT  = 4 * 8  * 32 * 16;
    if (idx < NODE_TOT) {
        int i    = idx & 15;
        int lane = (idx >> 4) & 31;
        int t    = idx >> 9;          // kt*24 + tn
        int tn   = t % 24;
        int kt   = t / 24;
        int col  = tn * 16 + (lane & 15);
        int k    = kt * 32 + krel_map(lane, i);
        wnode_frag[idx] = f2bf(Wnode[k * 384 + col]);
    } else if (idx < NODE_TOT + OUT_TOT) {
        int j    = idx - NODE_TOT;
        int i    = j & 15;
        int lane = (j >> 4) & 31;
        int t    = j >> 9;            // kt*8 + tn
        int tn   = t % 8;
        int kt   = t / 8;
        int col  = tn * 16 + (lane & 15);
        int k    = kt * 32 + krel_map(lane, i);
        wout_frag[j] = f2bf(Wout[k * 128 + col]);
    }
}

// ---- embeddings f32 -> bf16 -----------------------------------------------
__global__ void k_emb2bf(const float* __restrict__ emb, ushort_t* __restrict__ out) {
    size_t i = (size_t)blockIdx.x * blockDim.x + threadIdx.x;
    if (i < (size_t)BB * NN * DD) out[i] = f2bf(emb[i]);
}

// ---- graph embed (mean over N) + fixed_ctx = ge @ W_fixed -----------------
__global__ void k_fixedctx(const float* __restrict__ emb,
                           const float* __restrict__ Wfixed,
                           float* __restrict__ fixed_ctx) {
    int b = blockIdx.x, d = threadIdx.x;       // block = 128
    __shared__ float ge[DD];
    float s = 0.f;
    const float* row = emb + (size_t)b * NN * DD + d;
    for (int n = 0; n < NN; ++n) s += row[(size_t)n * DD];
    ge[d] = s * (1.0f / NN);
    __syncthreads();
    float acc = 0.f;
    for (int k = 0; k < DD; ++k) acc += ge[k] * Wfixed[k * DD + d];
    fixed_ctx[b * DD + d] = acc;
}

// ---- proj = emb_bf16 @ W_node (WMMA bf16), split -> Kbf/Vbf/Lbf -----------
// M = B*N = 131072, K = 128, Ncols = 384.  8 waves/block, 1 16x16 tile/wave.
__global__ void k_proj_wmma(const ushort_t* __restrict__ Abf,
                            const ushort_t* __restrict__ wnode_frag,
                            ushort_t* __restrict__ Kbf,
                            ushort_t* __restrict__ Vbf,
                            ushort_t* __restrict__ Lbf) {
    int wid  = threadIdx.x >> 5;
    int lane = threadIdx.x & 31;
    long tile = (long)blockIdx.x * 8 + wid;    // (M/16)*(384/16) = 196608 tiles
    int tm = (int)(tile / 24);
    int tn = (int)(tile % 24);

    v8f c = {};
    int rowA = tm * 16 + (lane & 15);
    const ushort_t* Arow = Abf + (size_t)rowA * DD;
    int off0 = (lane & 16) ? 8 : 0;

#pragma unroll
    for (int kt = 0; kt < 4; ++kt) {
        int kk = kt * 32;
        Frag32B a, bfrag;
        a.q[0] = *(const uint4*)(Arow + kk + off0);
        a.q[1] = *(const uint4*)(Arow + kk + off0 + 16);
        const ushort_t* bsrc = wnode_frag + ((size_t)((kt * 24 + tn) * 32 + lane)) * 16;
        bfrag.q[0] = *(const uint4*)(bsrc);
        bfrag.q[1] = *(const uint4*)(bsrc + 8);
        c = __builtin_amdgcn_wmma_f32_16x16x32_bf16(
                false, a.v, false, bfrag.v, (short)0, c, false, false);
    }

    int ncol  = tn * 16 + (lane & 15);         // 0..383
    int mbase = tm * 16 + ((lane & 16) ? 8 : 0);
    ushort_t* outp; int cc;
    if (ncol < 128)      { outp = Kbf; cc = ncol; }
    else if (ncol < 256) { outp = Vbf; cc = ncol - 128; }
    else                 { outp = Lbf; cc = ncol - 256; }
#pragma unroll
    for (int r = 0; r < 8; ++r) {
        int m = mbase + r;                      // global row = b*512 + n
        outp[(size_t)m * DD + cc] = f2bf(c[r]);
    }
}

// ---- per step: q = fixed_ctx + [emb[cur],frac] @ W_step -------------------
__global__ void k_stepq(const float* __restrict__ emb,
                        const float* __restrict__ Wstep,
                        const float* __restrict__ fixed_ctx,
                        const int* __restrict__ current,
                        float* __restrict__ qbuf, float frac) {
    int b = blockIdx.x, d = threadIdx.x;       // block = 128
    __shared__ float ce[DD];
    int cur = current[b];
    ce[d] = emb[((size_t)b * NN + cur) * DD + d];
    __syncthreads();
    float acc = fixed_ctx[b * DD + d];
    for (int k = 0; k < DD; ++k) acc += ce[k] * Wstep[k * DD + d];
    acc += frac * Wstep[DD * DD + d];          // last row of [D+1, D]
    qbuf[b * DD + d] = acc;
}

// ---- per step: fused masked multi-head attention (glimpse heads) ----------
// block = 256 threads (8 waves) per batch element b. wave w == head h.
__global__ void k_attn(const float* __restrict__ qbuf,
                       const ushort_t* __restrict__ Kbf,
                       const ushort_t* __restrict__ Vbf,
                       const unsigned char* __restrict__ visited,
                       ushort_t* __restrict__ heads_bf) {
    int b = blockIdx.x;
    int tid = threadIdx.x, wid = tid >> 5, lane = tid & 31;
    __shared__ float qs[DD];
    __shared__ float cs[HH * NN];              // 16 KB
    const float inv_sqrt_dk = 0.25f;           // 1/sqrt(16)

    if (tid < DD) qs[tid] = qbuf[b * DD + tid];
    __syncthreads();

    // compat[h][n] for 2 n per thread, all 8 heads
    for (int rep = 0; rep < 2; ++rep) {
        int n = tid + rep * 256;
        const uint4* kr = (const uint4*)(Kbf + ((size_t)b * NN + n) * DD);
        float acc[HH];
#pragma unroll
        for (int h = 0; h < HH; ++h) acc[h] = 0.f;
#pragma unroll
        for (int q4 = 0; q4 < 16; ++q4) {
            float f[8];
            unpack8(kr[q4], f);
            int d0 = q4 * 8;
#pragma unroll
            for (int j = 0; j < 8; ++j) acc[(d0 + j) >> 4] += qs[d0 + j] * f[j];
        }
        bool vis = visited[b * NN + n] != 0;
#pragma unroll
        for (int h = 0; h < HH; ++h)
            cs[h * NN + n] = vis ? NEGV : acc[h] * inv_sqrt_dk;
    }
    __syncthreads();

    // softmax + weighted sum of V, one wave per head; lane owns 16 n's
    int h = wid;
    float* crow = cs + h * NN;
    float lmax = -INFINITY;
#pragma unroll
    for (int i = 0; i < 16; ++i) lmax = fmaxf(lmax, crow[lane * 16 + i]);
#pragma unroll
    for (int off = 1; off < 32; off <<= 1) lmax = fmaxf(lmax, __shfl_xor(lmax, off, 32));
    float lsum = 0.f;
#pragma unroll
    for (int i = 0; i < 16; ++i) {
        float e = __expf(crow[lane * 16 + i] - lmax);
        crow[lane * 16 + i] = e;
        lsum += e;
    }
#pragma unroll
    for (int off = 1; off < 32; off <<= 1) lsum += __shfl_xor(lsum, off, 32);
    float rs = 1.f / lsum;

    float hv[DKV];
#pragma unroll
    for (int j = 0; j < DKV; ++j) hv[j] = 0.f;
    for (int i = 0; i < 16; ++i) {
        int n = lane * 16 + i;
        float w = crow[n];
        const uint4* vr = (const uint4*)(Vbf + ((size_t)b * NN + n) * DD + h * DKV);
        float f[8];
        unpack8(vr[0], f);
#pragma unroll
        for (int j = 0; j < 8; ++j) hv[j] += w * f[j];
        unpack8(vr[1], f);
#pragma unroll
        for (int j = 0; j < 8; ++j) hv[8 + j] += w * f[j];
    }
#pragma unroll
    for (int off = 1; off < 32; off <<= 1)
#pragma unroll
        for (int j = 0; j < DKV; ++j) hv[j] += __shfl_xor(hv[j], off, 32);
    if (lane == 0) {
#pragma unroll
        for (int j = 0; j < DKV; ++j)
            heads_bf[b * DD + h * DKV + j] = f2bf(hv[j] * rs);
    }
}

// ---- per step: glimpse = heads @ W_out (WMMA bf16) ------------------------
// M=256, K=128, N=128. 16 blocks x 8 waves, 1 tile per wave.
__global__ void k_glimpse_wmma(const ushort_t* __restrict__ heads_bf,
                               const ushort_t* __restrict__ wout_frag,
                               ushort_t* __restrict__ glimpse_bf) {
    int wid = threadIdx.x >> 5, lane = threadIdx.x & 31;
    int tm = blockIdx.x;                       // 0..15
    int tn = wid;                              // 0..7

    v8f c = {};
    int rowA = tm * 16 + (lane & 15);
    const ushort_t* Arow = heads_bf + (size_t)rowA * DD;
    int off0 = (lane & 16) ? 8 : 0;
#pragma unroll
    for (int kt = 0; kt < 4; ++kt) {
        int kk = kt * 32;
        Frag32B a, bfrag;
        a.q[0] = *(const uint4*)(Arow + kk + off0);
        a.q[1] = *(const uint4*)(Arow + kk + off0 + 16);
        const ushort_t* bsrc = wout_frag + ((size_t)((kt * 8 + tn) * 32 + lane)) * 16;
        bfrag.q[0] = *(const uint4*)(bsrc);
        bfrag.q[1] = *(const uint4*)(bsrc + 8);
        c = __builtin_amdgcn_wmma_f32_16x16x32_bf16(
                false, a.v, false, bfrag.v, (short)0, c, false, false);
    }
    int ncol  = tn * 16 + (lane & 15);
    int mbase = tm * 16 + ((lane & 16) ? 8 : 0);
#pragma unroll
    for (int r = 0; r < 8; ++r)
        glimpse_bf[(size_t)(mbase + r) * DD + ncol] = f2bf(c[r]);
}

// ---- per step: logits -> tanh clip -> mask -> log_softmax -> argmax -------
// block = 256 per batch element b.
__global__ void k_logits(const ushort_t* __restrict__ glimpse_bf,
                         const ushort_t* __restrict__ Lbf,
                         unsigned char* __restrict__ visited,
                         int* __restrict__ current,
                         float* __restrict__ logp_out,   // [B, T, N]
                         int* __restrict__ seq_out,      // [B, T]
                         int t_step, int T) {
    int b = blockIdx.x, tid = threadIdx.x, wid = tid >> 5, lane = tid & 31;
    __shared__ float gs[DD];
    __shared__ float lg[NN];
    __shared__ float wmax[8]; __shared__ int warg[8];
    __shared__ float wsum[8];
    __shared__ float s_gm, s_ls; __shared__ int s_ga;
    const float inv_sqrt_d = 0.08838834764831845f;   // 1/sqrt(128)

    if (tid < DD) gs[tid] = bf2f(glimpse_bf[b * DD + tid]);
    __syncthreads();

    for (int rep = 0; rep < 2; ++rep) {
        int n = tid + rep * 256;
        const uint4* lr = (const uint4*)(Lbf + ((size_t)b * NN + n) * DD);
        float acc = 0.f;
#pragma unroll
        for (int q4 = 0; q4 < 16; ++q4) {
            float f[8];
            unpack8(lr[q4], f);
            int d0 = q4 * 8;
#pragma unroll
            for (int j = 0; j < 8; ++j) acc += gs[d0 + j] * f[j];
        }
        float l = tanhf(acc * inv_sqrt_d) * CLIPV;
        if (visited[b * NN + n]) l = NEGV;
        lg[n] = l;
    }
    __syncthreads();

    // max + argmax (first-index tie break)
    float best = lg[tid]; int bi = tid;
    {
        float v2 = lg[tid + 256];
        if (v2 > best) { best = v2; bi = tid + 256; }
    }
#pragma unroll
    for (int off = 1; off < 32; off <<= 1) {
        float ov = __shfl_xor(best, off, 32);
        int   oi = __shfl_xor(bi, off, 32);
        if (ov > best || (ov == best && oi < bi)) { best = ov; bi = oi; }
    }
    if (lane == 0) { wmax[wid] = best; warg[wid] = bi; }
    __syncthreads();
    if (tid == 0) {
        float g = wmax[0]; int gi = warg[0];
        for (int w = 1; w < 8; ++w)
            if (wmax[w] > g || (wmax[w] == g && warg[w] < gi)) { g = wmax[w]; gi = warg[w]; }
        s_gm = g; s_ga = gi;
    }
    __syncthreads();
    float gm = s_gm;

    float psum = __expf(lg[tid] - gm) + __expf(lg[tid + 256] - gm);
#pragma unroll
    for (int off = 1; off < 32; off <<= 1) psum += __shfl_xor(psum, off, 32);
    if (lane == 0) wsum[wid] = psum;
    __syncthreads();
    if (tid == 0) {
        float s = 0.f;
        for (int w = 0; w < 8; ++w) s += wsum[w];
        s_ls = logf(s);
    }
    __syncthreads();
    float ls = s_ls;

    float* orow = logp_out + ((size_t)b * T + t_step) * NN;
    orow[tid]       = lg[tid] - gm - ls;
    orow[tid + 256] = lg[tid + 256] - gm - ls;

    if (tid == 0) {
        int sel = s_ga;
        seq_out[b * T + t_step] = sel;
        visited[b * NN + sel] = 1;
        current[b] = sel;
    }
}

// ---------------------------------------------------------------------------
extern "C" void kernel_launch(void* const* d_in, const int* in_sizes, int n_in,
                              void* d_out, int out_size, void* d_ws, size_t ws_size,
                              hipStream_t stream) {
    const float* emb    = (const float*)d_in[0];   // [B,N,D]
    const float* Wnode  = (const float*)d_in[1];   // [D,3D]
    const float* Wfixed = (const float*)d_in[2];   // [D,D]
    const float* Wstep  = (const float*)d_in[3];   // [D+1,D]
    const float* Wout   = (const float*)d_in[4];   // [D,D]
    (void)in_sizes; (void)n_in; (void)ws_size;

    int T = out_size / (BB * (NN + 1));            // B*T*N + B*T
    if (T <= 0) T = 64;

    // ---- workspace carve-up (all chunks 256B-aligned sizes) ----
    char* w = (char*)d_ws;
    const size_t EMB_ELEMS = (size_t)BB * NN * DD;
    ushort_t* emb_bf   = (ushort_t*)w;  w += EMB_ELEMS * 2;          // 33.5 MB
    ushort_t* Kbf      = (ushort_t*)w;  w += EMB_ELEMS * 2;          // 33.5 MB (L2-resident)
    ushort_t* Vbf      = (ushort_t*)w;  w += EMB_ELEMS * 2;
    ushort_t* Lbf      = (ushort_t*)w;  w += EMB_ELEMS * 2;
    ushort_t* wnode_fr = (ushort_t*)w;  w += (size_t)4 * 24 * 32 * 16 * 2;
    ushort_t* wout_fr  = (ushort_t*)w;  w += (size_t)4 * 8 * 32 * 16 * 2;
    float*    fixedctx = (float*)w;     w += (size_t)BB * DD * 4;
    float*    qbuf     = (float*)w;     w += (size_t)BB * DD * 4;
    ushort_t* heads_bf = (ushort_t*)w;  w += (size_t)BB * DD * 2 + 256;
    ushort_t* glim_bf  = (ushort_t*)w;  w += (size_t)BB * DD * 2 + 256;
    unsigned char* visited = (unsigned char*)w; w += (size_t)BB * NN;
    int*      current  = (int*)w;       w += (size_t)BB * 4;

    float* logp_out = (float*)d_out;
    int*   seq_out  = (int*)(logp_out + (size_t)BB * T * NN);

    // ---- precompute ----
    k_init<<<(BB * NN + 255) / 256, 256, 0, stream>>>(visited, current);
    k_weight_frags<<<(4 * 24 * 32 * 16 + 4 * 8 * 32 * 16 + 255) / 256, 256, 0, stream>>>(
        Wnode, Wout, wnode_fr, wout_fr);
    k_emb2bf<<<(int)((EMB_ELEMS + 255) / 256), 256, 0, stream>>>(emb, emb_bf);
    k_fixedctx<<<BB, DD, 0, stream>>>(emb, Wfixed, fixedctx);
    // 196608 wave-tiles / 8 waves per block
    k_proj_wmma<<<24576, 256, 0, stream>>>(emb_bf, wnode_fr, Kbf, Vbf, Lbf);

    // ---- autoregressive decode ----
    for (int t = 0; t < T; ++t) {
        float frac = (float)t / (float)T;
        k_stepq<<<BB, DD, 0, stream>>>(emb, Wstep, fixedctx, current, qbuf, frac);
        k_attn<<<BB, 256, 0, stream>>>(qbuf, Kbf, Vbf, visited, heads_bf);
        k_glimpse_wmma<<<16, 256, 0, stream>>>(heads_bf, wout_fr, glim_bf);
        k_logits<<<BB, 256, 0, stream>>>(glim_bf, Lbf, visited, current,
                                         logp_out, seq_out, t, T);
    }
}
